// SiameseClassifier_65816078844269
// MI455X (gfx1250) — matrix operand order repaired and checked by reading, not compiled, not run on hardware
//
#include <hip/hip_runtime.h>

typedef __attribute__((ext_vector_type(16))) _Float16 v16h;
typedef __attribute__((ext_vector_type(8)))  _Float16 v8h;
typedef __attribute__((ext_vector_type(8)))  float    v8f;

#define SS 2048
#define BB 64
#define DD 512
#define EE 512
#define CC 3
#define NCHUNK 4
#define SCHUNK (SS / NCHUNK)       // 512
#define APITCH 520                 // LDS row pitch in halves (bank-skewed)

// workspace layout (float offsets)
#define POOLS_OFF 0                              // 2*64*1536
#define PSUM_OFF  (2 * BB * 3 * EE)              // 2*4*64*512
#define PMAX_OFF  (PSUM_OFF + 2 * NCHUNK * BB * EE)
#define PLAST_OFF (PMAX_OFF + 2 * NCHUNK * BB * EE)

// CDNA5 hardware tanh (TRANS op). Two v_nops cover the ISA §7.4 trans-op
// result hazard since the compiler cannot schedule into an asm block.
__device__ __forceinline__ float tanh_hw(float x) {
    float r;
    asm volatile("v_tanh_f32 %0, %1\n\tv_nop\n\tv_nop" : "=v"(r) : "v"(x));
    return r;
}

// ---------------------------------------------------------------------------
// Kernel 1: fused  h = tanh(x @ W_enc)  + masked pooling, one S-chunk.
// Block = 16 b-rows x 128 e-cols; 8 waves = 8 e-tiles sharing one
// double-buffered f16 A-tile staged in LDS (converted once per block).
// A-fragment LDS reads are software-pipelined 4 ahead through a 5-slot
// rotating register file so WMMAs don't sit on s_wait_dscnt 0.
// ---------------------------------------------------------------------------
__global__ __launch_bounds__(256) void encode_pool_partial(
    const float* __restrict__ x1, const float* __restrict__ x2,
    const int*   __restrict__ len1, const int* __restrict__ len2,
    const float* __restrict__ Wenc, float* __restrict__ ws)
{
    const int b0    = blockIdx.x * 16;          // batch tile
    const int e0    = blockIdx.y * 128;         // embedding range
    const int z     = blockIdx.z / NCHUNK;      // which input tensor
    const int chunk = blockIdx.z % NCHUNK;      // S-chunk

    const float* x    = (z == 0) ? x1 : x2;
    const int*   lens = (z == 0) ? len1 : len2;

    const int tid  = threadIdx.x;
    const int wv   = tid >> 5;                  // wave = e-tile index
    const int lane = tid & 31;
    const int half = lane >> 4;                 // lane-half selects K sub-range
    const int lr   = lane & 15;

    __shared__ _Float16 As[2 * 16 * APITCH];    // double-buffered A tile (f16)

    // ---- Preload B fragments: W_enc[:, wave's e-tile], f32 -> f16 ----------
    v16h bfrag[16];
    {
        const int col = e0 + wv * 16 + lr;
        const int kh  = half * 16;
        #pragma unroll
        for (int kk = 0; kk < 16; ++kk) {
            const float* wp = Wenc + (size_t)(kk * 32 + kh) * EE + col;
            v16h bv;
            #pragma unroll
            for (int i = 0; i < 16; ++i)
                bv[i] = (_Float16)wp[(size_t)i * EE];
            bfrag[kk] = bv;
        }
    }

    // C-tile element owned by this thread for VGPR r:  b = b0 + r + 8*half
    int lenv[8];
    #pragma unroll
    for (int r = 0; r < 8; ++r) {
        int L = lens[b0 + r + 8 * half];
        lenv[r] = (L < 1) ? 1 : L;
    }

    float sumv[8], mxv[8], lastv[8];
    #pragma unroll
    for (int r = 0; r < 8; ++r) {
        sumv[r] = 0.0f; mxv[r] = -__builtin_inff(); lastv[r] = 0.0f;
    }

    // cooperative fill mapping: 256 threads cover 16 rows x 16 K-chunks of 32
    const int rowF = tid & 15;
    const int kcF  = (tid >> 4) * 32;
    const float* xrow = x + ((size_t)(b0 + rowF)) * DD + kcF;

    const int s_begin = chunk * SCHUNK;
    const int s_end   = s_begin + SCHUNK;

    float4 R[8];
    // prologue: stage A(s_begin) into buffer 0
    {
        const float* rp = xrow + (size_t)s_begin * BB * DD;
        #pragma unroll
        for (int j = 0; j < 8; ++j) R[j] = *(const float4*)(rp + j * 4);
        _Float16* dst = &As[rowF * APITCH + kcF];
        #pragma unroll
        for (int j = 0; j < 4; ++j) {
            float4 fa = R[2 * j], fb = R[2 * j + 1];
            v8h hx;
            hx[0]=(_Float16)fa.x; hx[1]=(_Float16)fa.y; hx[2]=(_Float16)fa.z; hx[3]=(_Float16)fa.w;
            hx[4]=(_Float16)fb.x; hx[5]=(_Float16)fb.y; hx[6]=(_Float16)fb.z; hx[7]=(_Float16)fb.w;
            *(v8h*)(dst + j * 8) = hx;
        }
    }
    __syncthreads();

    for (int s = s_begin; s < s_end; ++s) {
        const int p = (s - s_begin) & 1;
        const bool more = (s + 1 < s_end);

        // issue next s-step's global loads early; consumed after compute
        if (more) {
            const float* rp = xrow + (size_t)(s + 1) * BB * DD;
            #pragma unroll
            for (int j = 0; j < 8; ++j) R[j] = *(const float4*)(rp + j * 4);
        }

        // ---- compute from LDS buffer p: 16 WMMAs over K=512 ----------------
        const _Float16* abase = &As[p * 16 * APITCH + lr * APITCH];
        #define LDFRAG(kk_)  ({                                               \
            const int kb_ = (kk_) * 32 + half * 16;                           \
            v8h lo_ = *(const v8h*)(abase + kb_);                             \
            v8h hi_ = *(const v8h*)(abase + kb_ + 8);                         \
            v16h a_;                                                          \
            _Pragma("unroll")                                                 \
            for (int i_ = 0; i_ < 8; ++i_) { a_[i_] = lo_[i_]; a_[i_ + 8] = hi_[i_]; } \
            a_; })

        v16h af[5];
        af[0] = LDFRAG(0); af[1] = LDFRAG(1);
        af[2] = LDFRAG(2); af[3] = LDFRAG(3);

        v8f acc = {};
        #pragma unroll
        for (int kk = 0; kk < 16; ++kk) {
            acc = __builtin_amdgcn_wmma_f32_16x16x32_f16(
                false, af[kk % 5], false, bfrag[kk], (short)0, acc, false, false);
            if (kk + 4 < 16) af[(kk + 4) % 5] = LDFRAG(kk + 4);
        }
        #undef LDFRAG

        // tanh + masked pooling (registers only)
        #pragma unroll
        for (int r = 0; r < 8; ++r) {
            float h = tanh_hw(acc[r]);
            bool  m = (s < lenv[r]);
            sumv[r] += m ? h : 0.0f;
            mxv[r]   = m ? fmaxf(mxv[r], h) : mxv[r];
            if (s == SS - 1) lastv[r] = h;
        }

        // convert + store next tile into the other buffer
        if (more) {
            _Float16* dst = &As[(p ^ 1) * 16 * APITCH + rowF * APITCH + kcF];
            #pragma unroll
            for (int j = 0; j < 4; ++j) {
                float4 fa = R[2 * j], fb = R[2 * j + 1];
                v8h hx;
                hx[0]=(_Float16)fa.x; hx[1]=(_Float16)fa.y; hx[2]=(_Float16)fa.z; hx[3]=(_Float16)fa.w;
                hx[4]=(_Float16)fb.x; hx[5]=(_Float16)fb.y; hx[6]=(_Float16)fb.z; hx[7]=(_Float16)fb.w;
                *(v8h*)(dst + j * 8) = hx;
            }
        }
        __syncthreads();
    }

    // ---- write per-chunk partials (each wave owns distinct (b,e) elems) ----
    const int e = e0 + wv * 16 + lr;
    float* Ps = ws + PSUM_OFF  + (size_t)(z * NCHUNK + chunk) * BB * EE;
    float* Pm = ws + PMAX_OFF  + (size_t)(z * NCHUNK + chunk) * BB * EE;
    float* Pl = ws + PLAST_OFF + (size_t)z * BB * EE;
    #pragma unroll
    for (int r = 0; r < 8; ++r) {
        const int b = b0 + r + 8 * half;
        Ps[(size_t)b * EE + e] = sumv[r];
        Pm[(size_t)b * EE + e] = mxv[r];
        if (chunk == NCHUNK - 1) Pl[(size_t)b * EE + e] = lastv[r];
    }
}

// ---------------------------------------------------------------------------
// Kernel 2: deterministic cross-chunk combine -> pools[z][b][last|max|avg]
// ---------------------------------------------------------------------------
__global__ __launch_bounds__(256) void combine_pools(
    const int* __restrict__ len1, const int* __restrict__ len2,
    float* __restrict__ ws)
{
    const int idx = blockIdx.x * 256 + threadIdx.x;   // over 2*64*512
    const int z   = idx / (BB * EE);
    const int rem = idx - z * (BB * EE);
    const int b   = rem / EE;
    const int e   = rem - b * EE;

    float s = 0.0f, m = -__builtin_inff();
    #pragma unroll
    for (int c = 0; c < NCHUNK; ++c) {
        s += ws[PSUM_OFF + (size_t)(z * NCHUNK + c) * BB * EE + rem];
        m  = fmaxf(m, ws[PMAX_OFF + (size_t)(z * NCHUNK + c) * BB * EE + rem]);
    }
    float last = ws[PLAST_OFF + (size_t)z * BB * EE + rem];

    int L = (z == 0) ? len1[b] : len2[b];
    if (L < 1) L = 1;

    float* P = ws + POOLS_OFF + (size_t)z * (BB * 3 * EE) + (size_t)b * (3 * EE);
    P[e]          = last;
    P[EE + e]     = m;
    P[2 * EE + e] = s / (float)L;
}

// ---------------------------------------------------------------------------
// Kernel 3: features = [u, v, |u-v|, u*v]  ->  out = features @ W_lin + b_lin
// ---------------------------------------------------------------------------
__global__ __launch_bounds__(256) void classify_kernel(
    const float* __restrict__ ws, const float* __restrict__ Wlin,
    const float* __restrict__ blin, float* __restrict__ out)
{
    const int b   = blockIdx.x;
    const int tid = threadIdx.x;
    const float* U = ws + POOLS_OFF + (size_t)b * (3 * EE);
    const float* V = ws + POOLS_OFF + (size_t)BB * (3 * EE) + (size_t)b * (3 * EE);

    float a0 = 0.0f, a1 = 0.0f, a2 = 0.0f;
    for (int j = tid; j < 12 * EE; j += 256) {
        int region = j / (3 * EE);
        int k      = j - region * (3 * EE);
        float u = U[k], v = V[k];
        float f = (region == 0) ? u
                : (region == 1) ? v
                : (region == 2) ? fabsf(u - v)
                                : u * v;
        const float* wl = Wlin + (size_t)j * CC;
        a0 += f * wl[0]; a1 += f * wl[1]; a2 += f * wl[2];
    }

    __shared__ float red[256 * 3];
    red[tid * 3 + 0] = a0; red[tid * 3 + 1] = a1; red[tid * 3 + 2] = a2;
    __syncthreads();
    for (int off = 128; off > 0; off >>= 1) {
        if (tid < off) {
            red[tid * 3 + 0] += red[(tid + off) * 3 + 0];
            red[tid * 3 + 1] += red[(tid + off) * 3 + 1];
            red[tid * 3 + 2] += red[(tid + off) * 3 + 2];
        }
        __syncthreads();
    }
    if (tid < CC) out[b * CC + tid] = red[tid] + blin[tid];
}

extern "C" void kernel_launch(void* const* d_in, const int* in_sizes, int n_in,
                              void* d_out, int out_size, void* d_ws, size_t ws_size,
                              hipStream_t stream) {
    const float* in1  = (const float*)d_in[0];
    const float* in2  = (const float*)d_in[1];
    const int*   l1   = (const int*)  d_in[2];
    const int*   l2   = (const int*)  d_in[3];
    const float* Wenc = (const float*)d_in[4];
    const float* Wlin = (const float*)d_in[5];
    const float* blin = (const float*)d_in[6];
    float* out = (float*)d_out;
    float* ws  = (float*)d_ws;   // ~3 MB used

    dim3 g1(BB / 16, EE / 128, 2 * NCHUNK);
    encode_pool_partial<<<g1, 256, 0, stream>>>(in1, in2, l1, l2, Wenc, ws);
    combine_pools<<<(2 * BB * EE) / 256, 256, 0, stream>>>(l1, l2, ws);
    classify_kernel<<<BB, 256, 0, stream>>>(ws, Wlin, blin, out);
}
